// DMRGCNBlock_13898514170201
// MI455X (gfx1250) — matrix operand order repaired and checked by reading, not compiled, not run on hardware
//
#include <hip/hip_runtime.h>
#include <hip/hip_bf16.h>

// CDNA5 / gfx1250 WMMA vector types
typedef __attribute__((ext_vector_type(16))) __bf16 v16bf;
typedef __attribute__((ext_vector_type(8)))  float  v8f;

// B=64, R=2, T=32, N=128, C_IN=C_OUT=64
// d_out = [out (B*C*T*N floats) | A (B*R*T*N*N floats)]

// ---------------------------------------------------------------------------
// B-operand swizzled LDS layout: each 32x16 (KxN) tile stored fragment-major.
//   lane = ((k>>4)&1)*16 + (n&15), element e = k&15
//   addr = tileBase + lane*LSTR + e   (LSTR=24 bf16 = 48B: 16B-aligned per lane)
constexpr int LSTR  = 24;
constexpr int TILEB = 32 * LSTR;   // 768 bf16 per swizzled tile

// A operand (16x32 MxK, row-major source), contiguous k order used here (same as B tiles):
//   lanes 0-15: row=lane, K {0..15}; lanes 16-31: row=lane-16, K {16..31}
__device__ __forceinline__ v16bf frag_a(const __bf16* p, int ld, int lane) {
    const int row = lane & 15;
    const int hi  = lane >> 4;
    const __bf16* q = p + row * ld + hi * 16;
    v16bf a;
#pragma unroll
    for (int e = 0; e < 16; ++e) {
        a[e]     = q[e];
    }
    return a;
}
// B operand from swizzled tile: 16 contiguous bf16 per lane -> 2x ds_load_b128
__device__ __forceinline__ v16bf frag_b_sw(const __bf16* tile, int lane) {
    const __bf16* q = tile + lane * LSTR;
    v16bf b;
#pragma unroll
    for (int e = 0; e < 16; ++e) b[e] = q[e];
    return b;
}
// Store 4 consecutive-K elements (k0..k0+3, fixed n) into swizzled tile grid
// (8 N-tiles per K-tile row). Contiguous 4xbf16 -> ds_store_b64.
__device__ __forceinline__ void store_b_sw(__bf16* base, int k0, int n,
                                           float v0, float v1, float v2, float v3) {
    const int kt    = k0 >> 5;
    const int nt    = n >> 4;
    const int laneB = (((k0 >> 4) & 1) << 4) + (n & 15);
    __bf16* d = base + (kt * 8 + nt) * TILEB + laneB * LSTR + (k0 & 15);
    d[0] = (__bf16)v0; d[1] = (__bf16)v1; d[2] = (__bf16)v2; d[3] = (__bf16)v3;
}

// ---------------------------------------------------------------------------
// Kernel 1: per (b,t):
//   Y_r = W_r @ X + b_gcn ;  Anorm = D^-1/2 (A+I) D^-1/2 ;  Z = sum_r Y_r @ Anorm
__global__ __launch_bounds__(256)
void gcn_agg_kernel(const float* __restrict__ x, const float* __restrict__ A,
                    const float* __restrict__ wg, const float* __restrict__ bg,
                    float* __restrict__ xcomb) {
    constexpr int YS_LD = 136;
    __shared__ __bf16 Ws[128 * 64];        // [r*64+o][c] row-major (A-op)  16 KB
    __shared__ __bf16 Ss[16 * TILEB];      // X slab, B-swizzled            24 KB
    __shared__ __bf16 An[32 * TILEB];      // Anorm, B-swizzled             48 KB
    __shared__ __bf16 Ys[64 * YS_LD];      // Y, row-major (A-op)           17 KB
    __shared__ float  dinv[128];

    const int b    = blockIdx.x >> 5;
    const int t    = blockIdx.x & 31;
    const int tid  = threadIdx.x;
    const int wave = tid >> 5;
    const int lane = tid & 31;
    const int m    = wave & 3;     // M tile
    const int nset = wave >> 2;    // N tile half
    const int hi   = lane >> 4;
    const int ln   = lane & 15;

    // Stage W_gcn (flat copy, row-major == source layout)
    const float4* wg4 = (const float4*)wg;
    for (int i = tid; i < 2048; i += 256) {
        float4 v = wg4[i];
        int base = i << 2;
        Ws[base + 0] = (__bf16)v.x; Ws[base + 1] = (__bf16)v.y;
        Ws[base + 2] = (__bf16)v.z; Ws[base + 3] = (__bf16)v.w;
    }
    // Stage X slab into B-swizzled tiles: thread = (n, c0/4); coalesced over n
    for (int i = tid; i < 2048; i += 256) {
        int n  = i & 127;
        int c0 = (i >> 7) << 2;
        float v[4];
#pragma unroll
        for (int j = 0; j < 4; ++j)
            v[j] = x[((((size_t)b * 64 + c0 + j) * 32 + t) << 7) + n];
        store_b_sw(Ss, c0, n, v[0], v[1], v[2], v[3]);
    }
    __syncthreads();

    v8f zacc[4] = {};

    for (int r = 0; r < 2; ++r) {
        const float* Ab = A + ((((size_t)b * 2 + r) * 32 + t) << 14);
        // --- degree rowsums -> rsqrt ---
        {
            int row = tid >> 1, half = tid & 1;
            const float4* ar = (const float4*)(Ab + (row << 7) + (half << 6));
            float s = 0.f;
#pragma unroll
            for (int i = 0; i < 16; ++i) { float4 v = ar[i]; s += v.x + v.y + v.z + v.w; }
            s += __shfl_xor(s, 1, 32);
            if (half == 0) dinv[row] = rsqrtf(1.0f + s);
            if (r == 0) __builtin_prefetch(Ab + (1 << 14) + (tid << 6), 0, 3);
        }
        __syncthreads();
        // --- normalized adjacency -> B-swizzled bf16 LDS ---
        for (int i = tid; i < 4096; i += 256) {
            int n  = i & 127;
            int k0 = (i >> 7) << 2;
            float dn = dinv[n];
            float v[4];
#pragma unroll
            for (int j = 0; j < 4; ++j) {
                int k = k0 + j;
                float a = Ab[((size_t)k << 7) + n];
                v[j] = (a + (k == n ? 1.f : 0.f)) * dinv[k] * dn;
            }
            store_b_sw(An, k0, n, v[0], v[1], v[2], v[3]);
        }
        // --- Y_r = W_r @ X (+ bias), K = 64 ---
#pragma unroll
        for (int nt = 0; nt < 4; ++nt) {
            int ntile = (nset << 2) + nt;
            v8f acc = {};
#pragma unroll
            for (int kt = 0; kt < 2; ++kt) {
                v16bf af = frag_a(&Ws[(r * 64 + (m << 4)) * 64 + kt * 32], 64, lane);
                v16bf bf = frag_b_sw(&Ss[((kt << 3) + ntile) * TILEB], lane);
                acc = __builtin_amdgcn_wmma_f32_16x16x32_bf16(
                    false, af, false, bf, (short)0, acc, false, false);
            }
#pragma unroll
            for (int j = 0; j < 8; ++j) {
                int o = (m << 4) + (hi << 3) + j;      // C/D: M = j + hi*8
                float y = acc[j] + bg[r * 64 + o];
                Ys[o * YS_LD + (ntile << 4) + ln] = (__bf16)y;
            }
        }
        __syncthreads();
        // --- Z += Y_r @ Anorm, K = 128 ---
#pragma unroll
        for (int nt = 0; nt < 4; ++nt) {
            int ntile = (nset << 2) + nt;
#pragma unroll
            for (int kt = 0; kt < 4; ++kt) {
                v16bf af = frag_a(&Ys[(m << 4) * YS_LD + kt * 32], YS_LD, lane);
                v16bf bf = frag_b_sw(&An[((kt << 3) + ntile) * TILEB], lane);
                zacc[nt] = __builtin_amdgcn_wmma_f32_16x16x32_bf16(
                    false, af, false, bf, (short)0, zacc[nt], false, false);
            }
        }
        __syncthreads();
    }
    // --- store x_comb slab (L2-resident scratch) ---
#pragma unroll
    for (int nt = 0; nt < 4; ++nt) {
        int ncol = ((nset << 2) + nt) << 4;
#pragma unroll
        for (int j = 0; j < 8; ++j) {
            int o = (m << 4) + (hi << 3) + j;
            xcomb[((((size_t)b * 64 + o) * 32 + t) << 7) + ncol + ln] = zacc[nt][j];
        }
    }
}

// ---------------------------------------------------------------------------
// Kernel 2: temporal conv (k=3, pad 1) as 64x192 @ 192x128 GEMM per (b,t),
// fused PReLU(a_tcn) on input, + b_tcn + residual + PReLU(a_out).
__global__ __launch_bounds__(256)
void tcn_kernel(const float* __restrict__ xcomb, const float* __restrict__ x,
                const float* __restrict__ wt, const float* __restrict__ bt,
                const float* __restrict__ a_tcn_p, const float* __restrict__ a_out_p,
                float* __restrict__ out) {
    constexpr int WT_LD = 200;
    __shared__ __bf16 Hs[48 * TILEB];     // H_cat (192x128), B-swizzled   72 KB
    __shared__ __bf16 Wt[64 * WT_LD];     // W_cat row-major (A-op)        25 KB

    const int b    = blockIdx.x >> 5;
    const int t    = blockIdx.x & 31;
    const int tid  = threadIdx.x;
    const int wave = tid >> 5;
    const int lane = tid & 31;
    const int m    = wave & 3;
    const int nset = wave >> 2;
    const int hi   = lane >> 4;
    const int ln   = lane & 15;

    const float a_t = a_tcn_p[0];
    const float a_o = a_out_p[0];

    // Stage W_cat[o][kk*64 + i] = w_tcn[o][i][kk]
    for (int i = tid; i < 64 * 192; i += 256) {
        int o = i / 192, k = i - o * 192;
        int kk = k >> 6, ic = k & 63;
        Wt[o * WT_LD + k] = (__bf16)wt[(o * 64 + ic) * 3 + kk];
    }
    // Stage H_cat = PReLU(x_comb) for t-1,t,t+1 into B-swizzled tiles.
    // thread = (n, k0/4): global loads coalesced over n, LDS stores b64.
    for (int i = tid; i < 6144; i += 256) {
        int n  = i & 127;
        int k0 = (i >> 7) << 2;             // 0..188, same kk block for all 4
        int kk = k0 >> 6;
        int c0 = k0 & 63;
        int ts = t + kk - 1;
        float v[4] = {0.f, 0.f, 0.f, 0.f};
        if (ts >= 0 && ts < 32) {
#pragma unroll
            for (int j = 0; j < 4; ++j) {
                float h = xcomb[((((size_t)b * 64 + c0 + j) * 32 + ts) << 7) + n];
                v[j] = h >= 0.f ? h : a_t * h;
            }
        }
        store_b_sw(Hs, k0, n, v[0], v[1], v[2], v[3]);
    }
    __syncthreads();

#pragma unroll
    for (int nt = 0; nt < 4; ++nt) {
        int ntile = (nset << 2) + nt;
        v8f acc = {};
#pragma unroll
        for (int kt = 0; kt < 6; ++kt) {   // K = 192
            v16bf af = frag_a(&Wt[(m << 4) * WT_LD + kt * 32], WT_LD, lane);
            v16bf bf = frag_b_sw(&Hs[((kt << 3) + ntile) * TILEB], lane);
            acc = __builtin_amdgcn_wmma_f32_16x16x32_bf16(
                false, af, false, bf, (short)0, acc, false, false);
        }
#pragma unroll
        for (int j = 0; j < 8; ++j) {
            int o = (m << 4) + (hi << 3) + j;
            size_t gi = ((((size_t)b * 64 + o) * 32 + t) << 7) + (ntile << 4) + ln;
            float v = acc[j] + bt[o] + x[gi];   // bias + residual
            out[gi] = v >= 0.f ? v : a_o * v;   // final PReLU
        }
    }
}

// ---------------------------------------------------------------------------
extern "C" void kernel_launch(void* const* d_in, const int* in_sizes, int n_in,
                              void* d_out, int out_size, void* d_ws, size_t ws_size,
                              hipStream_t stream) {
    const float* x     = (const float*)d_in[0];
    const float* A     = (const float*)d_in[1];
    const float* wg    = (const float*)d_in[2];
    const float* bg    = (const float*)d_in[3];
    const float* wt    = (const float*)d_in[4];
    const float* bt    = (const float*)d_in[5];
    const float* a_tcn = (const float*)d_in[6];
    const float* a_out = (const float*)d_in[7];
    float* out   = (float*)d_out;
    float* xcomb = (float*)d_ws;   // B*C*T*N floats = 64 MB scratch

    dim3 grid(64 * 32), block(256);
    gcn_agg_kernel<<<grid, block, 0, stream>>>(x, A, wg, bg, xcomb);
    tcn_kernel<<<grid, block, 0, stream>>>(xcomb, x, wt, bt, a_tcn, a_out, out);

    const size_t out0 = (size_t)64 * 64 * 32 * 128;
    hipMemcpyAsync(out + out0, A, (size_t)in_sizes[1] * sizeof(float),
                   hipMemcpyDeviceToDevice, stream);
}
